// MemoryEnhancedRNN_84799834293134
// MI455X (gfx1250) — matrix operand order repaired and checked by reading, not compiled, not run on hardware
//
#include <hip/hip_runtime.h>
#include <math.h>

typedef __bf16 bf16_t;
typedef __attribute__((ext_vector_type(16))) __bf16 v16bf;
typedef __attribute__((ext_vector_type(8)))  __bf16 v8bf;
typedef __attribute__((ext_vector_type(8)))  float  v8f;

#define Bb 32
#define Ss 128
#define Ii 512
#define Hh 1024
#define H3 3072
#define Nn 16384
#define Ww 128
#define CW (Hh + Ww) /* 1152 */
#define GRU_BLOCKS 32   /* 16 blocks layer0 + 16 blocks layer1, 4 waves each */

__device__ __forceinline__ float sigmf_(float x){ return 1.f / (1.f + expf(-x)); }
__device__ __forceinline__ float softplusf_(float x){ return (x > 20.f) ? x : log1pf(expf(x)); }

__device__ __forceinline__ v8f wmma_bf16(v16bf a, v16bf b, v8f c){
  // D = A(16x32 bf16) * B(32x16 bf16) + C(16x16 f32)
  return __builtin_amdgcn_wmma_f32_16x16x32_bf16(false, a, false, b, (short)0, c, false, false);
}

// A fragment: 16x32 bf16. lane<16: row = mbase+lane, K = {k0..k0+7, k0+16..k0+23}
//             lane>=16:   row = mbase+(lane-16), K = {k0+8..k0+15, k0+24..k0+31}
__device__ __forceinline__ v16bf load_frag_a(const bf16_t* __restrict__ A, int lda,
                                             int mbase, int k0, int lane){
  int m  = mbase + (lane & 15);
  int kb = k0 + ((lane & 16) ? 8 : 0);
  const bf16_t* p = A + (size_t)m * lda + kb;
  v8bf lo = *(const v8bf*)(p);
  v8bf hi = *(const v8bf*)(p + 16);
  v16bf f;
#pragma unroll
  for (int i = 0; i < 8; ++i){ f[i] = lo[i]; f[8 + i] = hi[i]; }
  return f;
}

// B fragment: 32x16 bf16 (KxN). col = nbase+(lane&15), K order same as A fragment.
__device__ __forceinline__ v16bf load_frag_b(const bf16_t* __restrict__ Wt, int ldn,
                                             int nbase, int k0, int lane){
  int n  = nbase + (lane & 15);
  int kb = k0 + ((lane & 16) ? 8 : 0);
  const bf16_t* p = Wt + (size_t)kb * ldn + n;
  v16bf f;
#pragma unroll
  for (int i = 0; i < 8; ++i){ f[i] = p[(size_t)i * ldn]; f[8 + i] = p[(size_t)(16 + i) * ldn]; }
  return f;
}

// ---------------- conversion / init kernels ----------------

__global__ void f32_to_bf16_kernel(const float* __restrict__ src, bf16_t* __restrict__ dst, int n){
  int i = blockIdx.x * blockDim.x + threadIdx.x;
  if (i < n) dst[i] = (bf16_t)src[i];
}

// W: [Nr][Kc] fp32  ->  Wt: [Kc][Nr] bf16
__global__ void transpose_to_bf16_kernel(const float* __restrict__ W, bf16_t* __restrict__ Wt,
                                         int Nr, int Kc){
  int i = blockIdx.x * blockDim.x + threadIdx.x;
  if (i < Nr * Kc){
    int r = i / Kc, c = i % Kc;
    Wt[(size_t)c * Nr + r] = (bf16_t)W[i];
  }
}

__global__ void zero_f32_kernel(float* p, int n){
  int i = blockIdx.x * blockDim.x + threadIdx.x;
  if (i < n) p[i] = 0.f;
}
__global__ void zero_bf16_kernel(bf16_t* p, int n){
  int i = blockIdx.x * blockDim.x + threadIdx.x;
  if (i < n) p[i] = (bf16_t)0.f;
}
__global__ void zero_i32_kernel(int* p, int n){
  int i = blockIdx.x * blockDim.x + threadIdx.x;
  if (i < n) p[i] = 0;
}

// ---------------- device-scope grid barrier (persistent kernel) ----------------
__device__ __forceinline__ void grid_sync(int* bar_count, int* bar_gen){
  __syncthreads();
  if (threadIdx.x == 0){
    __threadfence();  // publish this block's writes device-wide
    int g = __hip_atomic_load(bar_gen, __ATOMIC_ACQUIRE, __HIP_MEMORY_SCOPE_AGENT);
    int a = __hip_atomic_fetch_add(bar_count, 1, __ATOMIC_ACQ_REL, __HIP_MEMORY_SCOPE_AGENT);
    if (a == GRU_BLOCKS - 1){
      __hip_atomic_store(bar_count, 0, __ATOMIC_RELAXED, __HIP_MEMORY_SCOPE_AGENT);
      __hip_atomic_fetch_add(bar_gen, 1, __ATOMIC_ACQ_REL, __HIP_MEMORY_SCOPE_AGENT);
    } else {
      while (__hip_atomic_load(bar_gen, __ATOMIC_ACQUIRE, __HIP_MEMORY_SCOPE_AGENT) == g){
        __builtin_amdgcn_s_sleep(2);
      }
    }
  }
  __syncthreads();
}

// ---------------- one GRU step for one 16-column tile (both GEMMs + 3 gates) --------
// 8 v8f accumulators: r,z fused (x@Wih + h@Whh), inn (x only), hn (h only),
// each for two 16-row M tiles (batch = 32 rows).
__device__ __forceinline__ void gru_tile(const bf16_t* __restrict__ Xb, int ldx, int Kin,
                                         const bf16_t* __restrict__ Hb,
                                         const float*  __restrict__ Hf,
                                         const bf16_t* __restrict__ WtIH,  // [Kin][3H]
                                         const bf16_t* __restrict__ WtHH,  // [H][3H]
                                         const float* __restrict__ bih,
                                         const float* __restrict__ bhh,
                                         float* __restrict__ HfN,
                                         bf16_t* __restrict__ HbN,
                                         int j0, int lane)
{
  v8f aR0 = {}, aR1 = {}, aZ0 = {}, aZ1 = {}, aI0 = {}, aI1 = {}, aN0 = {}, aN1 = {};

  for (int k0 = 0; k0 < Kin; k0 += 32){
    v16bf x0 = load_frag_a(Xb, ldx, 0,  k0, lane);
    v16bf x1 = load_frag_a(Xb, ldx, 16, k0, lane);
    v16bf bR = load_frag_b(WtIH, H3, j0,          k0, lane);
    v16bf bZ = load_frag_b(WtIH, H3, Hh + j0,     k0, lane);
    v16bf bN = load_frag_b(WtIH, H3, 2 * Hh + j0, k0, lane);
    aR0 = wmma_bf16(x0, bR, aR0);  aR1 = wmma_bf16(x1, bR, aR1);
    aZ0 = wmma_bf16(x0, bZ, aZ0);  aZ1 = wmma_bf16(x1, bZ, aZ1);
    aI0 = wmma_bf16(x0, bN, aI0);  aI1 = wmma_bf16(x1, bN, aI1);
  }
  for (int k0 = 0; k0 < Hh; k0 += 32){
    v16bf h0 = load_frag_a(Hb, Hh, 0,  k0, lane);
    v16bf h1 = load_frag_a(Hb, Hh, 16, k0, lane);
    v16bf bR = load_frag_b(WtHH, H3, j0,          k0, lane);
    v16bf bZ = load_frag_b(WtHH, H3, Hh + j0,     k0, lane);
    v16bf bN = load_frag_b(WtHH, H3, 2 * Hh + j0, k0, lane);
    aR0 = wmma_bf16(h0, bR, aR0);  aR1 = wmma_bf16(h1, bR, aR1);
    aZ0 = wmma_bf16(h0, bZ, aZ0);  aZ1 = wmma_bf16(h1, bZ, aZ1);
    aN0 = wmma_bf16(h0, bN, aN0);  aN1 = wmma_bf16(h1, bN, aN1);
  }

  const int n  = j0 + (lane & 15);
  const float biasR = bih[n] + bhh[n];
  const float biasZ = bih[Hh + n] + bhh[Hh + n];
  const float biasI = bih[2 * Hh + n];
  const float biasN = bhh[2 * Hh + n];
  const int mo = 8 * (lane >> 4);

#pragma unroll
  for (int v = 0; v < 8; ++v){
    {
      int m = v + mo;
      float r  = sigmf_(aR0[v] + biasR);
      float z  = sigmf_(aZ0[v] + biasZ);
      float nn = tanhf(aI0[v] + biasI + r * (aN0[v] + biasN));
      float ho = Hf[m * Hh + n];
      float h2 = (1.f - z) * nn + z * ho;
      HfN[m * Hh + n] = h2;
      HbN[m * Hh + n] = (bf16_t)h2;
    }
    {
      int m = 16 + v + mo;
      float r  = sigmf_(aR1[v] + biasR);
      float z  = sigmf_(aZ1[v] + biasZ);
      float nn = tanhf(aI1[v] + biasI + r * (aN1[v] + biasN));
      float ho = Hf[m * Hh + n];
      float h2 = (1.f - z) * nn + z * ho;
      HfN[m * Hh + n] = h2;
      HbN[m * Hh + n] = (bf16_t)h2;
    }
  }
}

// ---------------- persistent 2-layer GRU: 129 pipelined slots, 1 launch ----------------
// Blocks 0..15 own layer-0 column tiles, blocks 16..31 own layer-1 tiles.
// Slot s: layer0 computes step s (s<128), layer1 computes step s-1 (s>=1).
__global__ void __launch_bounds__(128) gru_persistent_kernel(
    const bf16_t* __restrict__ xbf,
    const bf16_t* __restrict__ WtIH0, const bf16_t* __restrict__ WtHH0,
    const bf16_t* __restrict__ WtIH1, const bf16_t* __restrict__ WtHH1,
    const float* __restrict__ bih0, const float* __restrict__ bhh0,
    const float* __restrict__ bih1, const float* __restrict__ bhh1,
    float* h0f0, float* h0f1, bf16_t* h0b0, bf16_t* h0b1,
    float* h1f0, float* h1f1, bf16_t* h1b0, bf16_t* h1b1,
    int* bar_count, int* bar_gen)
{
  const int lane = threadIdx.x & 31;
  const int wave = threadIdx.x >> 5;
  const bool is_l1 = blockIdx.x >= 16;
  const int blk = is_l1 ? (blockIdx.x - 16) : blockIdx.x;
  const int j0 = (blk * 4 + wave) * 16;

  float*  h0f[2] = { h0f0, h0f1 };
  bf16_t* h0b[2] = { h0b0, h0b1 };
  float*  h1f[2] = { h1f0, h1f1 };
  bf16_t* h1b[2] = { h1b0, h1b1 };

  for (int slot = 0; slot <= Ss; ++slot){
    if (!is_l1){
      if (slot < Ss){
        const int rp = slot & 1, wp = rp ^ 1;
        gru_tile(xbf + (size_t)slot * Ii, Ss * Ii, Ii,
                 h0b[rp], h0f[rp], WtIH0, WtHH0, bih0, bhh0,
                 h0f[wp], h0b[wp], j0, lane);
      }
    } else {
      if (slot >= 1){
        const int t = slot - 1;
        const int rp = t & 1, wp = rp ^ 1;
        gru_tile(h0b[slot & 1], Hh, Hh,          // layer0 output of step t (parity (t+1)&1 == slot&1)
                 h1b[rp], h1f[rp], WtIH1, WtHH1, bih1, bhh1,
                 h1f[wp], h1b[wp], j0, lane);
      }
    }
    grid_sync(bar_count, bar_gen);
  }
  // final h1 = step 127 -> write parity (127+1)&1 = 0 : h1f0 / h1b0
}

// ---------------- generic 32-row WMMA GEMM: Out(32xN) = act(A(32xK) @ Wt(KxN) + bias)
__global__ void wmma_gemm32_kernel(const bf16_t* __restrict__ A, int lda, int K,
                                   const bf16_t* __restrict__ Wt, int Ncols,
                                   const float* __restrict__ bias,
                                   float* __restrict__ Out, int act)
{
  const int lane = threadIdx.x & 31;
  const int wave = threadIdx.x >> 5;
  const int tile = blockIdx.x * (blockDim.x >> 5) + wave;
  const int j0 = tile * 16;
  if (j0 >= Ncols) return;

  v8f c0 = {}, c1 = {};
  for (int k0 = 0; k0 < K; k0 += 32){
    v16bf a0 = load_frag_a(A, lda, 0,  k0, lane);
    v16bf a1 = load_frag_a(A, lda, 16, k0, lane);
    v16bf b  = load_frag_b(Wt, Ncols, j0, k0, lane);
    c0 = wmma_bf16(a0, b, c0);
    c1 = wmma_bf16(a1, b, c1);
  }
  const int n = j0 + (lane & 15);
  const float bs = bias[n];
  const int mo = 8 * (lane >> 4);
#pragma unroll
  for (int v = 0; v < 8; ++v){
    float x0 = c0[v] + bs, x1 = c1[v] + bs;
    if (act == 1){ x0 = sigmf_(x0); x1 = sigmf_(x1); }
    else if (act == 2){ x0 = tanhf(x0); x1 = tanhf(x1); }
    Out[(v + mo) * Ncols + n]      = x0;
    Out[(16 + v + mo) * Ncols + n] = x1;
  }
}

// ---------------- scalar heads: beta, g, gamma ----------------
__global__ void scalar_heads_kernel(const float* __restrict__ Hf,
                                    const float* __restrict__ Wbeta, const float* __restrict__ bbeta,
                                    const float* __restrict__ Wg,    const float* __restrict__ bg,
                                    const float* __restrict__ Wgam,  const float* __restrict__ bgam,
                                    float* beta, float* g, float* gamma)
{
  const int b = blockIdx.x, tid = threadIdx.x;
  float s1 = 0.f, s2 = 0.f, s3 = 0.f;
  for (int i = tid; i < Hh; i += 256){
    float h = Hf[b * Hh + i];
    s1 += h * Wbeta[i]; s2 += h * Wg[i]; s3 += h * Wgam[i];
  }
  __shared__ float r1[256], r2[256], r3[256];
  r1[tid] = s1; r2[tid] = s2; r3[tid] = s3; __syncthreads();
  for (int s = 128; s > 0; s >>= 1){
    if (tid < s){ r1[tid] += r1[tid + s]; r2[tid] += r2[tid + s]; r3[tid] += r3[tid + s]; }
    __syncthreads();
  }
  if (tid == 0){
    beta[b]  = softplusf_(r1[0] + bbeta[0]);
    g[b]     = sigmf_(r2[0] + bg[0]);
    gamma[b] = 1.f + softplusf_(r3[0] + bgam[0]);
  }
}

// normalize k rows in place: k /= (||k|| + eps)
__global__ void knorm_kernel(float* __restrict__ k){
  const int b = blockIdx.x, i = threadIdx.x; // 128 threads
  float v = k[b * Ww + i];
  __shared__ float red[128];
  red[i] = v * v; __syncthreads();
  for (int s = 64; s > 0; s >>= 1){ if (i < s) red[i] += red[i + s]; __syncthreads(); }
  k[b * Ww + i] = v / (sqrtf(red[0]) + 1e-8f);
}

// sim pass: wbuf[b][n] = beta[b] * (M[b,n,:] . k_n[b,:]) / (||M[b,n,:]|| + eps)
__global__ void sim_kernel(const float* __restrict__ M, const float* __restrict__ kn,
                           const float* __restrict__ beta, float* __restrict__ wbuf)
{
  const int b = blockIdx.x >> 11;      // 2048 chunks per batch
  const int chunk = blockIdx.x & 2047;
  const int wave = threadIdx.x >> 5, lane = threadIdx.x & 31;
  __shared__ float sk[128];
  if (threadIdx.x < 128) sk[threadIdx.x] = kn[b * Ww + threadIdx.x];
  __syncthreads();
  const int n = chunk * 8 + wave;
  const float* row = M + ((size_t)b * Nn + n) * Ww + lane * 4;
  __builtin_prefetch(row + 8 * Ww, 0, 1);
  float4 m4 = *(const float4*)row;
  float k0 = sk[lane * 4 + 0], k1 = sk[lane * 4 + 1], k2 = sk[lane * 4 + 2], k3 = sk[lane * 4 + 3];
  float dot = m4.x * k0 + m4.y * k1 + m4.z * k2 + m4.w * k3;
  float nrm = m4.x * m4.x + m4.y * m4.y + m4.z * m4.z + m4.w * m4.w;
  for (int off = 16; off > 0; off >>= 1){
    dot += __shfl_xor(dot, off, 32);
    nrm += __shfl_xor(nrm, off, 32);
  }
  if (lane == 0) wbuf[(size_t)b * Nn + n] = beta[b] * dot / (sqrtf(nrm) + 1e-8f);
}

// per-batch reductions over 16384 elements (one block / batch, 256 threads)
__global__ void reduce_max_wpsum_kernel(const float* __restrict__ wbuf, const float* __restrict__ wprev,
                                        float* maxbuf, float* wpsum){
  const int b = blockIdx.x, tid = threadIdx.x;
  float mx = -1e30f, sp = 0.f;
  for (int i = tid; i < Nn; i += 256){
    mx = fmaxf(mx, wbuf[(size_t)b * Nn + i]);
    sp += wprev[(size_t)b * Nn + i];
  }
  __shared__ float rm[256], rs[256];
  rm[tid] = mx; rs[tid] = sp; __syncthreads();
  for (int s = 128; s > 0; s >>= 1){
    if (tid < s){ rm[tid] = fmaxf(rm[tid], rm[tid + s]); rs[tid] += rs[tid + s]; }
    __syncthreads();
  }
  if (tid == 0){ maxbuf[b] = rm[0]; wpsum[b] = rs[0]; }
}

__global__ void exp_sum_kernel(float* __restrict__ wbuf, const float* __restrict__ maxbuf,
                               float* sumbuf){
  const int b = blockIdx.x, tid = threadIdx.x;
  const float mx = maxbuf[b];
  float s = 0.f;
  for (int i = tid; i < Nn; i += 256){
    float e = expf(wbuf[(size_t)b * Nn + i] - mx);
    wbuf[(size_t)b * Nn + i] = e;
    s += e;
  }
  __shared__ float rs[256];
  rs[tid] = s; __syncthreads();
  for (int st = 128; st > 0; st >>= 1){ if (tid < st) rs[tid] += rs[tid + st]; __syncthreads(); }
  if (tid == 0) sumbuf[b] = rs[0];
}

__global__ void mix_pow_kernel(float* __restrict__ wbuf, const float* __restrict__ wprev,
                               const float* __restrict__ sumbuf, const float* __restrict__ wpsum,
                               const float* __restrict__ g, const float* __restrict__ gamma,
                               float* powsum){
  const int b = blockIdx.x, tid = threadIdx.x;
  const float inv_sum = 1.f / sumbuf[b];
  const float inv_wps = 1.f / (wpsum[b] + 1e-8f);
  const float gb = g[b], gam = gamma[b];
  float s = 0.f;
  for (int i = tid; i < Nn; i += 256){
    float wc = wbuf[(size_t)b * Nn + i] * inv_sum;
    float wp = wprev[(size_t)b * Nn + i] * inv_wps;
    float w  = gb * wc + (1.f - gb) * wp;
    float v  = powf(w + 1e-8f, gam);
    wbuf[(size_t)b * Nn + i] = v;
    s += v;
  }
  __shared__ float rs[256];
  rs[tid] = s; __syncthreads();
  for (int st = 128; st > 0; st >>= 1){ if (tid < st) rs[tid] += rs[tid + st]; __syncthreads(); }
  if (tid == 0) powsum[b] = rs[0];
}

__global__ void norm_s2_kernel(float* __restrict__ wbuf, const float* __restrict__ powsum,
                               float* s2buf){
  const int b = blockIdx.x, tid = threadIdx.x;
  const float inv = 1.f / powsum[b];
  float s = 0.f;
  for (int i = tid; i < Nn; i += 256){
    float wn = wbuf[(size_t)b * Nn + i] * inv;
    wbuf[(size_t)b * Nn + i] = wn;
    s += wn * wn;
  }
  __shared__ float rs[256];
  rs[tid] = s; __syncthreads();
  for (int st = 128; st > 0; st >>= 1){ if (tid < st) rs[tid] += rs[tid + st]; __syncthreads(); }
  if (tid == 0) s2buf[b] = rs[0];
}

// second streaming pass over M: S1 = sum_n w*M, S2 = sum_n w^2*M
__global__ void accum_kernel(const float* __restrict__ M, const float* __restrict__ wbuf,
                             float* __restrict__ S1, float* __restrict__ S2){
  const int b = blockIdx.x >> 5, chunk = blockIdx.x & 31;  // 32 chunks x 512 rows
  const int wave = threadIdx.x >> 5, lane = threadIdx.x & 31;
  float a1x = 0, a1y = 0, a1z = 0, a1w = 0;
  float a2x = 0, a2y = 0, a2z = 0, a2w = 0;
  const int nbase = chunk * 512 + wave * 64;
  for (int r = 0; r < 64; ++r){
    const int n = nbase + r;
    const float wn = wbuf[(size_t)b * Nn + n];
    const float* row = M + ((size_t)b * Nn + n) * Ww + lane * 4;
    __builtin_prefetch(row + 4 * Ww, 0, 1);
    float4 m4 = *(const float4*)row;
    const float wn2 = wn * wn;
    a1x += wn * m4.x;  a1y += wn * m4.y;  a1z += wn * m4.z;  a1w += wn * m4.w;
    a2x += wn2 * m4.x; a2y += wn2 * m4.y; a2z += wn2 * m4.z; a2w += wn2 * m4.w;
  }
  __shared__ float sS1[128], sS2[128];
  if (threadIdx.x < 128){ sS1[threadIdx.x] = 0.f; sS2[threadIdx.x] = 0.f; }
  __syncthreads();
  atomicAdd(&sS1[lane * 4 + 0], a1x); atomicAdd(&sS1[lane * 4 + 1], a1y);
  atomicAdd(&sS1[lane * 4 + 2], a1z); atomicAdd(&sS1[lane * 4 + 3], a1w);
  atomicAdd(&sS2[lane * 4 + 0], a2x); atomicAdd(&sS2[lane * 4 + 1], a2y);
  atomicAdd(&sS2[lane * 4 + 2], a2z); atomicAdd(&sS2[lane * 4 + 3], a2w);
  __syncthreads();
  if (threadIdx.x < 128){
    atomicAdd(&S1[b * Ww + threadIdx.x], sS1[threadIdx.x]);
    atomicAdd(&S2[b * Ww + threadIdx.x], sS2[threadIdx.x]);
  }
}

// combined = [h(bf16) , bf16(r)] where r = S1 - e*S2 + s2*a
__global__ void comb_kernel(const bf16_t* __restrict__ hb,
                            const float* __restrict__ S1, const float* __restrict__ S2,
                            const float* __restrict__ e, const float* __restrict__ a,
                            const float* __restrict__ s2buf,
                            bf16_t* __restrict__ comb){
  int idx = blockIdx.x * blockDim.x + threadIdx.x;
  if (idx >= Bb * CW) return;
  int b = idx / CW, c = idx % CW;
  bf16_t v;
  if (c < Hh) v = hb[b * Hh + c];
  else {
    int i = c - Hh;
    float r = S1[b * Ww + i] - e[b * Ww + i] * S2[b * Ww + i] + s2buf[b] * a[b * Ww + i];
    v = (bf16_t)r;
  }
  comb[idx] = v;
}

// ------------------------------------------------------------------

extern "C" void kernel_launch(void* const* d_in, const int* in_sizes, int n_in,
                              void* d_out, int out_size, void* d_ws, size_t ws_size,
                              hipStream_t stream) {
  const float* x      = (const float*)d_in[0];
  const float* memory = (const float*)d_in[1];
  const float* w_prev = (const float*)d_in[2];
  const float* W_ih0  = (const float*)d_in[3];
  const float* W_hh0  = (const float*)d_in[4];
  const float* b_ih0  = (const float*)d_in[5];
  const float* b_hh0  = (const float*)d_in[6];
  const float* W_ih1  = (const float*)d_in[7];
  const float* W_hh1  = (const float*)d_in[8];
  const float* b_ih1  = (const float*)d_in[9];
  const float* b_hh1  = (const float*)d_in[10];
  const float* Wk     = (const float*)d_in[11];
  const float* bk     = (const float*)d_in[12];
  const float* Wbeta  = (const float*)d_in[13];
  const float* bbeta  = (const float*)d_in[14];
  const float* Wg     = (const float*)d_in[15];
  const float* bg     = (const float*)d_in[16];
  const float* Wgamma = (const float*)d_in[17];
  const float* bgamma = (const float*)d_in[18];
  const float* We     = (const float*)d_in[19];
  const float* be     = (const float*)d_in[20];
  const float* Wa     = (const float*)d_in[21];
  const float* ba     = (const float*)d_in[22];
  const float* Wout   = (const float*)d_in[23];
  const float* bout   = (const float*)d_in[24];
  float* out = (float*)d_out;

  char* ws = (char*)d_ws;
  size_t off = 0;
  auto alloc = [&](size_t bytes) -> void* {
    void* p = ws + off;
    off = (off + bytes + 255) & ~(size_t)255;
    return p;
  };

  bf16_t* xbf    = (bf16_t*)alloc((size_t)Bb * Ss * Ii * 2);
  bf16_t* WtIH0  = (bf16_t*)alloc((size_t)Ii * H3 * 2);
  bf16_t* WtHH0  = (bf16_t*)alloc((size_t)Hh * H3 * 2);
  bf16_t* WtIH1  = (bf16_t*)alloc((size_t)Hh * H3 * 2);
  bf16_t* WtHH1  = (bf16_t*)alloc((size_t)Hh * H3 * 2);
  bf16_t* WtK    = (bf16_t*)alloc((size_t)Hh * Ww * 2);
  bf16_t* WtE    = (bf16_t*)alloc((size_t)Hh * Ww * 2);
  bf16_t* WtA    = (bf16_t*)alloc((size_t)Hh * Ww * 2);
  bf16_t* WtOut  = (bf16_t*)alloc((size_t)CW * Ii * 2);
  float*  h0f_a  = (float*)alloc(Bb * Hh * 4);
  float*  h0f_b  = (float*)alloc(Bb * Hh * 4);
  float*  h1f_a  = (float*)alloc(Bb * Hh * 4);
  float*  h1f_b  = (float*)alloc(Bb * Hh * 4);
  bf16_t* h0b_a  = (bf16_t*)alloc(Bb * Hh * 2);
  bf16_t* h0b_b  = (bf16_t*)alloc(Bb * Hh * 2);
  bf16_t* h1b_a  = (bf16_t*)alloc(Bb * Hh * 2);
  bf16_t* h1b_b  = (bf16_t*)alloc(Bb * Hh * 2);
  float*  kbuf   = (float*)alloc(Bb * Ww * 4);
  float*  ebuf   = (float*)alloc(Bb * Ww * 4);
  float*  abuf   = (float*)alloc(Bb * Ww * 4);
  float*  beta   = (float*)alloc(Bb * 4);
  float*  gbuf   = (float*)alloc(Bb * 4);
  float*  gamma  = (float*)alloc(Bb * 4);
  float*  maxbuf = (float*)alloc(Bb * 4);
  float*  wpsum  = (float*)alloc(Bb * 4);
  float*  sumbuf = (float*)alloc(Bb * 4);
  float*  powsum = (float*)alloc(Bb * 4);
  float*  s2buf  = (float*)alloc(Bb * 4);
  int*    bar    = (int*)alloc(2 * sizeof(int));     // [0]=count, [1]=gen
  float*  S1     = (float*)alloc(Bb * Ww * 4);       // S1,S2 contiguous (256B multiples)
  float*  S2     = (float*)alloc(Bb * Ww * 4);
  float*  wbuf   = (float*)alloc((size_t)Bb * Nn * 4);
  bf16_t* comb   = (bf16_t*)alloc(Bb * CW * 2);
  (void)ws_size; (void)in_sizes; (void)n_in; (void)out_size;

  // --- weight / input conversion (bf16, transposed to [K][N] where needed) ---
  {
    int n = Bb * Ss * Ii;
    f32_to_bf16_kernel<<<(n + 255) / 256, 256, 0, stream>>>(x, xbf, n);
    transpose_to_bf16_kernel<<<(H3 * Ii + 255) / 256, 256, 0, stream>>>(W_ih0, WtIH0, H3, Ii);
    transpose_to_bf16_kernel<<<(H3 * Hh + 255) / 256, 256, 0, stream>>>(W_hh0, WtHH0, H3, Hh);
    transpose_to_bf16_kernel<<<(H3 * Hh + 255) / 256, 256, 0, stream>>>(W_ih1, WtIH1, H3, Hh);
    transpose_to_bf16_kernel<<<(H3 * Hh + 255) / 256, 256, 0, stream>>>(W_hh1, WtHH1, H3, Hh);
    f32_to_bf16_kernel<<<(Hh * Ww + 255) / 256, 256, 0, stream>>>(Wk, WtK, Hh * Ww);
    f32_to_bf16_kernel<<<(Hh * Ww + 255) / 256, 256, 0, stream>>>(We, WtE, Hh * Ww);
    f32_to_bf16_kernel<<<(Hh * Ww + 255) / 256, 256, 0, stream>>>(Wa, WtA, Hh * Ww);
    transpose_to_bf16_kernel<<<(Ii * CW + 255) / 256, 256, 0, stream>>>(Wout, WtOut, Ii, CW);
  }

  // --- zero initial hidden state, accumulators, grid-barrier state ---
  zero_f32_kernel<<<(Bb * Hh + 255) / 256, 256, 0, stream>>>(h0f_a, Bb * Hh);
  zero_f32_kernel<<<(Bb * Hh + 255) / 256, 256, 0, stream>>>(h1f_a, Bb * Hh);
  zero_bf16_kernel<<<(Bb * Hh + 255) / 256, 256, 0, stream>>>(h0b_a, Bb * Hh);
  zero_bf16_kernel<<<(Bb * Hh + 255) / 256, 256, 0, stream>>>(h1b_a, Bb * Hh);
  zero_f32_kernel<<<(2 * Bb * Ww + 255) / 256, 256, 0, stream>>>(S1, 2 * Bb * Ww);
  zero_i32_kernel<<<1, 32, 0, stream>>>(bar, 2);

  // --- 2-layer GRU, single persistent launch, 129 pipelined slots ---
  // parity 0 buffers are the initial (zero) state; final h1 lands in parity 0.
  gru_persistent_kernel<<<GRU_BLOCKS, 128, 0, stream>>>(
      xbf, WtIH0, WtHH0, WtIH1, WtHH1,
      b_ih0, b_hh0, b_ih1, b_hh1,
      h0f_a, h0f_b, h0b_a, h0b_b,
      h1f_a, h1f_b, h1b_a, h1b_b,
      bar + 0, bar + 1);
  float*  hF = h1f_a;   // final h, fp32
  bf16_t* hB = h1b_a;   // final h, bf16

  // --- head GEMMs: k (linear), e (sigmoid), a (tanh) ---
  wmma_gemm32_kernel<<<2, 128, 0, stream>>>(hB, Hh, Hh, WtK, Ww, bk, kbuf, 0);
  wmma_gemm32_kernel<<<2, 128, 0, stream>>>(hB, Hh, Hh, WtE, Ww, be, ebuf, 1);
  wmma_gemm32_kernel<<<2, 128, 0, stream>>>(hB, Hh, Hh, WtA, Ww, ba, abuf, 2);
  scalar_heads_kernel<<<Bb, 256, 0, stream>>>(hF, Wbeta, bbeta, Wg, bg, Wgamma, bgamma,
                                              beta, gbuf, gamma);
  knorm_kernel<<<Bb, 128, 0, stream>>>(kbuf);

  // --- addressing: pass 1 over memory (cosine sim * beta) ---
  sim_kernel<<<Bb * 2048, 256, 0, stream>>>(memory, kbuf, beta, wbuf);
  reduce_max_wpsum_kernel<<<Bb, 256, 0, stream>>>(wbuf, w_prev, maxbuf, wpsum);
  exp_sum_kernel<<<Bb, 256, 0, stream>>>(wbuf, maxbuf, sumbuf);
  mix_pow_kernel<<<Bb, 256, 0, stream>>>(wbuf, w_prev, sumbuf, wpsum, gbuf, gamma, powsum);
  norm_s2_kernel<<<Bb, 256, 0, stream>>>(wbuf, powsum, s2buf);

  // --- pass 2 over memory: S1 = w.M, S2 = w^2.M  (M_new never materialized) ---
  accum_kernel<<<Bb * 32, 256, 0, stream>>>(memory, wbuf, S1, S2);

  // --- combined = [h, r],  r = S1 - e*S2 + (sum w^2)*a ---
  comb_kernel<<<(Bb * CW + 255) / 256, 256, 0, stream>>>(hB, S1, S2, ebuf, abuf, s2buf, comb);

  // --- out = combined @ Wout^T + bout ---
  wmma_gemm32_kernel<<<8, 128, 0, stream>>>(comb, CW, CW, WtOut, Ii, bout, out, 0);
}
